// SNNFCNet_20942260535416
// MI455X (gfx1250) — matrix operand order repaired and checked by reading, not compiled, not run on hardware
//
#include <hip/hip_runtime.h>
#include <hip/hip_bf16.h>

typedef __bf16 bf16_t;
typedef __attribute__((ext_vector_type(16))) __bf16 v16bf;
typedef __attribute__((ext_vector_type(8)))  __bf16 v8bf;
typedef __attribute__((ext_vector_type(8)))  float  v8f;

#define B_SZ   1024
#define NI     4096
#define NH     4096
#define NO     1024
#define TSTEPS 10
#define BETA_F   0.95f
#define THRESH_F 1.0f

// LDS tile row stride (elements). 40*2B = 80B = 5*16B: keeps ds_load_b128
// alignment while spreading the 16 row-lanes across distinct bank groups.
#define LDS_STRIDE 40

union Frag { v16bf v; v8bf h[2]; };

__device__ __forceinline__ v8f wmma_bf16(v16bf a, v16bf b, v8f c) {
    // D = A(16x32 bf16) * B(32x16 bf16) + C(16x16 f32)
    return __builtin_amdgcn_wmma_f32_16x16x32_bf16(
        /*neg_a=*/false, a, /*neg_b=*/false, b,
        /*c_mod=*/(short)0, c, /*reuse_a=*/false, /*reuse_b=*/false);
}

// Async global -> LDS 16-byte copy (per-lane), tracked by ASYNCcnt.
__device__ __forceinline__ void async_copy_b128(unsigned lds_off, const bf16_t* gptr) {
    asm volatile("global_load_async_to_lds_b128 %0, %1, off"
                 :: "v"(lds_off), "v"(gptr)
                 : "memory");
}

// ---------------------------------------------------------------------------
// fp32 -> (bf16 hi, bf16 lo) split: hi = bf16(f), lo = bf16(f - f32(hi))
// ---------------------------------------------------------------------------
__global__ __launch_bounds__(256)
void split_bf16_kernel(const float* __restrict__ src,
                       bf16_t* __restrict__ hi, bf16_t* __restrict__ lo,
                       size_t n) {
    size_t i = (size_t)blockIdx.x * blockDim.x + threadIdx.x;
    if (i >= n) return;
    float f = src[i];
    bf16_t h = (bf16_t)f;
    hi[i] = h;
    lo[i] = (bf16_t)(f - (float)h);
}

// ---------------------------------------------------------------------------
// LIF layer 1 (elementwise): reset from OLD mem, decay+input, fire on NEW mem.
// ---------------------------------------------------------------------------
__global__ __launch_bounds__(256)
void lif1_kernel(const float* __restrict__ cur1, float* __restrict__ mem1,
                 bf16_t* __restrict__ spk1, size_t n) {
    size_t i = (size_t)blockIdx.x * blockDim.x + threadIdx.x;
    if (i >= n) return;
    float m = mem1[i];
    float reset = (m - THRESH_F > 0.0f) ? 1.0f : 0.0f;
    m = BETA_F * m + cur1[i] - reset * THRESH_F;
    mem1[i] = m;
    spk1[i] = (bf16_t)((m - THRESH_F > 0.0f) ? 1.0f : 0.0f);
}

// ---------------------------------------------------------------------------
// GEMM1: cur1[M=1024, N=4096] = x[M, K=4096] @ W1[N, K]^T + b1
// Split-precision bf16: acc += Ah*Bh + Ah*Bl + Al*Bh.
// Block tile 128x128, 8 waves (4 along M x 2 along N), wave tile 32x64.
// B tiles (shared by all 4 M-waves) are staged in double-buffered LDS via
// global_load_async_to_lds_b128 (ASYNCcnt) and read back with ds_load_b128.
// A fragments load directly from global (only 2 waves share each A row).
// ---------------------------------------------------------------------------
__global__ __launch_bounds__(256)
void gemm1_wmma_kernel(const bf16_t* __restrict__ Ahi, const bf16_t* __restrict__ Alo,
                       const bf16_t* __restrict__ Whi, const bf16_t* __restrict__ Wlo,
                       const float* __restrict__ bias, float* __restrict__ C) {
    constexpr int K = NI;
    constexpr int N = NH;
    constexpr int KSTEPS = K / 32;

    // [buf][part(hi,lo)][128 rows * LDS_STRIDE] : 2*2*128*40*2B = 40 KB
    __shared__ bf16_t sB[2][2][128 * LDS_STRIDE];

    const int tid  = threadIdx.x;
    const int lane = tid & 31;
    const int wid  = tid >> 5;             // 0..7
    const int wm   = wid & 3;              // 4 waves along M
    const int wn   = wid >> 2;             // 2 waves along N
    const int m0   = blockIdx.y * 128 + wm * 32;
    const int n0b  = blockIdx.x * 128;     // block's N origin (W1 row)
    const int r    = lane & 15;
    const int h    = lane >> 4;

    const v8f vzero = {0.f, 0.f, 0.f, 0.f, 0.f, 0.f, 0.f, 0.f};
    v8f acc[2][4];
#pragma unroll
    for (int mi = 0; mi < 2; ++mi)
#pragma unroll
        for (int ni = 0; ni < 4; ++ni) acc[mi][ni] = vzero;

    size_t aoff[2];
#pragma unroll
    for (int mi = 0; mi < 2; ++mi)
        aoff[mi] = (size_t)(m0 + mi * 16 + r) * K + (size_t)(h * 8);

    // Issue one stage of async B-tile copies: 2 parts * 2 chunks = 4 per thread.
    auto issue_stage = [&](int kt, int buf) {
        const int k = kt * 32;
#pragma unroll
        for (int part = 0; part < 2; ++part) {
            const bf16_t* src = part ? Wlo : Whi;
#pragma unroll
            for (int c = 0; c < 2; ++c) {
                const int chunk = tid + c * 256;        // 0..511
                const int row = chunk >> 2;             // 0..127 (W1 row in tile)
                const int seg = chunk & 3;              // 16B segment in 64B row
                const bf16_t* g = src + (size_t)(n0b + row) * K + k + seg * 8;
                unsigned l = (unsigned)(uintptr_t)&sB[buf][part][row * LDS_STRIDE + seg * 8];
                async_copy_b128(l, g);
            }
        }
    };

    issue_stage(0, 0);

    for (int kt = 0; kt < KSTEPS; ++kt) {
        const int buf = kt & 1;
        const int k = kt * 32;

        if (kt + 1 < KSTEPS) {
            issue_stage(kt + 1, buf ^ 1);
            asm volatile("s_wait_asynccnt 0x4" ::: "memory");  // current stage done
        } else {
            asm volatile("s_wait_asynccnt 0x0" ::: "memory");
        }
        __syncthreads();

        // A fragments: direct from global.
        Frag ahi[2], alo[2];
#pragma unroll
        for (int mi = 0; mi < 2; ++mi) {
            const bf16_t* p = Ahi + aoff[mi] + k;
            const bf16_t* q = Alo + aoff[mi] + k;
            ahi[mi].h[0] = *(const v8bf*)(p);
            ahi[mi].h[1] = *(const v8bf*)(p + 16);
            alo[mi].h[0] = *(const v8bf*)(q);
            alo[mi].h[1] = *(const v8bf*)(q + 16);
        }

        // B fragments: from LDS (ds_load_b128 x2 per fragment).
        Frag bhi[4], blo[4];
#pragma unroll
        for (int ni = 0; ni < 4; ++ni) {
            const int row = wn * 64 + ni * 16 + r;      // W1 row within 128-tile
            const bf16_t* p = &sB[buf][0][row * LDS_STRIDE + h * 8];
            const bf16_t* q = &sB[buf][1][row * LDS_STRIDE + h * 8];
            bhi[ni].h[0] = *(const v8bf*)(p);
            bhi[ni].h[1] = *(const v8bf*)(p + 16);
            blo[ni].h[0] = *(const v8bf*)(q);
            blo[ni].h[1] = *(const v8bf*)(q + 16);
        }

#pragma unroll
        for (int mi = 0; mi < 2; ++mi)
#pragma unroll
            for (int ni = 0; ni < 4; ++ni) {
                acc[mi][ni] = wmma_bf16(ahi[mi].v, bhi[ni].v, acc[mi][ni]);
                acc[mi][ni] = wmma_bf16(ahi[mi].v, blo[ni].v, acc[mi][ni]);
                acc[mi][ni] = wmma_bf16(alo[mi].v, bhi[ni].v, acc[mi][ni]);
            }

        __syncthreads();  // all waves done reading buf before it is refilled
    }

#pragma unroll
    for (int mi = 0; mi < 2; ++mi)
#pragma unroll
        for (int ni = 0; ni < 4; ++ni) {
            const int n = n0b + wn * 64 + ni * 16 + r;
            const float bv = bias[n];
#pragma unroll
            for (int rr = 0; rr < 8; ++rr) {
                const int m = m0 + mi * 16 + h * 8 + rr;
                C[(size_t)m * N + n] = acc[mi][ni][rr] + bv;
            }
        }
}

// ---------------------------------------------------------------------------
// GEMM2 + fused LIF2: cur2[M=1024, N=1024] = spk1[M, K=4096] @ W2[N, K]^T + b2
// spk1 is exact in bf16 (0/1), so only W2 needs hi+lo split (2 WMMAs/tile).
// Direct global fragment loads (pipelined by the scheduler against WMMA).
// ---------------------------------------------------------------------------
__global__ __launch_bounds__(128)
void gemm2_lif2_kernel(const bf16_t* __restrict__ S,
                       const bf16_t* __restrict__ Whi, const bf16_t* __restrict__ Wlo,
                       const float* __restrict__ bias, float* __restrict__ mem2,
                       float* __restrict__ out_spk, float* __restrict__ out_mem) {
    constexpr int K = NH;
    constexpr int N = NO;
    const int lane = threadIdx.x & 31;
    const int wid  = threadIdx.x >> 5;     // 0..3
    const int wm   = wid & 1;
    const int wn   = wid >> 1;
    const int m0   = blockIdx.y * 64 + wm * 32;
    const int n0   = blockIdx.x * 64 + wn * 32;
    const int r    = lane & 15;
    const int h    = lane >> 4;

    const v8f vzero = {0.f, 0.f, 0.f, 0.f, 0.f, 0.f, 0.f, 0.f};
    v8f acc[2][2];
#pragma unroll
    for (int mi = 0; mi < 2; ++mi)
#pragma unroll
        for (int ni = 0; ni < 2; ++ni) acc[mi][ni] = vzero;

    size_t aoff[2], boff[2];
#pragma unroll
    for (int mi = 0; mi < 2; ++mi)
        aoff[mi] = (size_t)(m0 + mi * 16 + r) * K + (size_t)(h * 8);
#pragma unroll
    for (int ni = 0; ni < 2; ++ni)
        boff[ni] = (size_t)(n0 + ni * 16 + r) * K + (size_t)(h * 8);

    for (int k = 0; k < K; k += 32) {
        if (k + 32 < K) {
            __builtin_prefetch(Whi + boff[0] + k + 32, 0, 0);
            __builtin_prefetch(S + aoff[0] + k + 32, 0, 0);
        }
        Frag a[2];
#pragma unroll
        for (int mi = 0; mi < 2; ++mi) {
            const bf16_t* p = S + aoff[mi] + k;
            a[mi].h[0] = *(const v8bf*)(p);
            a[mi].h[1] = *(const v8bf*)(p + 16);
        }
        Frag bhi[2], blo[2];
#pragma unroll
        for (int ni = 0; ni < 2; ++ni) {
            const bf16_t* p = Whi + boff[ni] + k;
            const bf16_t* q = Wlo + boff[ni] + k;
            bhi[ni].h[0] = *(const v8bf*)(p);
            bhi[ni].h[1] = *(const v8bf*)(p + 16);
            blo[ni].h[0] = *(const v8bf*)(q);
            blo[ni].h[1] = *(const v8bf*)(q + 16);
        }
#pragma unroll
        for (int mi = 0; mi < 2; ++mi)
#pragma unroll
            for (int ni = 0; ni < 2; ++ni) {
                acc[mi][ni] = wmma_bf16(a[mi].v, bhi[ni].v, acc[mi][ni]);
                acc[mi][ni] = wmma_bf16(a[mi].v, blo[ni].v, acc[mi][ni]);
            }
    }

#pragma unroll
    for (int mi = 0; mi < 2; ++mi)
#pragma unroll
        for (int ni = 0; ni < 2; ++ni) {
            const int n = n0 + ni * 16 + r;
            const float bv = bias[n];
#pragma unroll
            for (int rr = 0; rr < 8; ++rr) {
                const int m = m0 + mi * 16 + h * 8 + rr;
                const size_t idx = (size_t)m * N + n;
                const float cur2 = acc[mi][ni][rr] + bv;
                const float mo = mem2[idx];
                const float reset = (mo - THRESH_F > 0.0f) ? 1.0f : 0.0f;
                const float mn = BETA_F * mo + cur2 - reset * THRESH_F;
                mem2[idx] = mn;
                out_spk[idx] = (mn - THRESH_F > 0.0f) ? 1.0f : 0.0f;
                out_mem[idx] = mn;
            }
        }
}

// ---------------------------------------------------------------------------
extern "C" void kernel_launch(void* const* d_in, const int* in_sizes, int n_in,
                              void* d_out, int out_size, void* d_ws, size_t ws_size,
                              hipStream_t stream) {
    const float* x  = (const float*)d_in[0];  // [1024, 4096]
    const float* W1 = (const float*)d_in[1];  // [4096, 4096]
    const float* b1 = (const float*)d_in[2];  // [4096]
    const float* W2 = (const float*)d_in[3];  // [1024, 4096]
    const float* b2 = (const float*)d_in[4];  // [1024]

    char* ws = (char*)d_ws;
    size_t off = 0;
    auto alloc = [&](size_t bytes) -> void* {
        void* p = (void*)(ws + off);
        off += (bytes + 255) & ~(size_t)255;
        return p;
    };

    bf16_t* xhi   = (bf16_t*)alloc((size_t)B_SZ * NI * sizeof(bf16_t));
    bf16_t* xlo   = (bf16_t*)alloc((size_t)B_SZ * NI * sizeof(bf16_t));
    bf16_t* W1hi  = (bf16_t*)alloc((size_t)NH * NI * sizeof(bf16_t));
    bf16_t* W1lo  = (bf16_t*)alloc((size_t)NH * NI * sizeof(bf16_t));
    bf16_t* W2hi  = (bf16_t*)alloc((size_t)NO * NH * sizeof(bf16_t));
    bf16_t* W2lo  = (bf16_t*)alloc((size_t)NO * NH * sizeof(bf16_t));
    float*  cur1  = (float*)alloc((size_t)B_SZ * NH * sizeof(float));
    float*  mem1  = (float*)alloc((size_t)B_SZ * NH * sizeof(float));
    float*  mem2  = (float*)alloc((size_t)B_SZ * NO * sizeof(float));
    bf16_t* spk1  = (bf16_t*)alloc((size_t)B_SZ * NH * sizeof(bf16_t));
    if (off > ws_size) return;  // workspace too small: cannot proceed

    float* out_spk_base = (float*)d_out;                                   // [T,B,NO]
    float* out_mem_base = (float*)d_out + (size_t)TSTEPS * B_SZ * NO;      // [T,B,NO]

    // Zero membrane states (ws is poisoned, must init every call).
    hipMemsetAsync(mem1, 0, (size_t)B_SZ * NH * sizeof(float), stream);
    hipMemsetAsync(mem2, 0, (size_t)B_SZ * NO * sizeof(float), stream);

    // Split fp32 operands into bf16 hi/lo pairs.
    {
        size_t nx = (size_t)B_SZ * NI;
        split_bf16_kernel<<<dim3((unsigned)(nx / 256)), dim3(256), 0, stream>>>(x, xhi, xlo, nx);
        size_t n1 = (size_t)NH * NI;
        split_bf16_kernel<<<dim3((unsigned)(n1 / 256)), dim3(256), 0, stream>>>(W1, W1hi, W1lo, n1);
        size_t n2 = (size_t)NO * NH;
        split_bf16_kernel<<<dim3((unsigned)(n2 / 256)), dim3(256), 0, stream>>>(W2, W2hi, W2lo, n2);
    }

    // cur1 = x @ W1^T + b1 (step-invariant, computed once).
    gemm1_wmma_kernel<<<dim3(NH / 128, B_SZ / 128), dim3(256), 0, stream>>>(
        xhi, xlo, W1hi, W1lo, b1, cur1);

    // Time loop: LIF1 -> (GEMM2 + LIF2 fused).
    for (int t = 0; t < TSTEPS; ++t) {
        size_t n1 = (size_t)B_SZ * NH;
        lif1_kernel<<<dim3((unsigned)(n1 / 256)), dim3(256), 0, stream>>>(cur1, mem1, spk1, n1);
        gemm2_lif2_kernel<<<dim3(NO / 64, B_SZ / 64), dim3(128), 0, stream>>>(
            spk1, W2hi, W2lo, b2, mem2,
            out_spk_base + (size_t)t * B_SZ * NO,
            out_mem_base + (size_t)t * B_SZ * NO);
    }
}
